// CoLA_69526930588313
// MI455X (gfx1250) — matrix operand order repaired and checked by reading, not compiled, not run on hardware
//
#include <hip/hip_runtime.h>

// ---------------- problem constants ----------------
#define B_    32
#define T_    750
#define F_    2048
#define EMB_  2048
#define C_    20
#define KEASY 150      // T/5
#define KHARD 37       // T/20
#define TPAD  784      // padded time axis for xT (index tp = t+1, borders zero)
#define KSLAB 128      // K-slab staged in LDS per step
#define NSLABS (F_ / KSLAB)

#define SMEM_A (3 * 128 * KSLAB * 2)          // 3 taps x 128 rows x KSLAB bf16
#define SMEM_B (3 * KSLAB * 32 * 2)           // 3 shift-aligned x-slabs [KSLAB][32]
#define SMEM_BUF (SMEM_A + SMEM_B)            // 122880 bytes per buffer
#define SMEM_TOTAL (2 * SMEM_BUF)             // double buffered: 245760 bytes

typedef __bf16        v16bf __attribute__((ext_vector_type(16)));
typedef float         v8f   __attribute__((ext_vector_type(8)));
typedef unsigned int  v8u   __attribute__((ext_vector_type(8)));
typedef unsigned int  v4u   __attribute__((ext_vector_type(4)));

#if __has_builtin(__builtin_amdgcn_tensor_load_to_lds) && __has_builtin(__builtin_amdgcn_s_wait_tensorcnt)
#define USE_TDM 1
#else
#define USE_TDM 0
#endif

__device__ __forceinline__ unsigned short f2bf(float f) {
  unsigned u = __float_as_uint(f);
  unsigned r = 0x7FFFu + ((u >> 16) & 1u);     // round-to-nearest-even
  return (unsigned short)((u + r) >> 16);
}

#if USE_TDM
typedef unsigned int v4u_ __attribute__((ext_vector_type(4)));
typedef int          v8i_ __attribute__((ext_vector_type(8)));
typedef int          v4i_ __attribute__((ext_vector_type(4)));

// Issue one 2-D TDM tile load: global (row-major, stride0 elements between
// rows) -> LDS (tile packed row-major). D# layout per CDNA5 ISA ch.8:
//   group0: count=1 | lds_addr[63:32] | global_addr[120:64] | type=2
//   group1: data_size=2B, tensor_dim0/1, tile_dim0/1, tensor_dim0_stride
// This toolchain's builtin takes 6 args (4 D# groups + spare group + cpol).
__device__ __forceinline__ void tdm_load_2d(
    unsigned lds_off, const void* gaddr,
    unsigned tile0, unsigned tile1,
    unsigned tensor0, unsigned tensor1,
    unsigned long long stride0_elems)
{
  unsigned long long ga = (unsigned long long)gaddr;
  v4u_ g0;
  g0[0] = 1u;                                           // count=1, user desc
  g0[1] = lds_off;                                      // lds_addr (bytes)
  g0[2] = (unsigned)(ga & 0xFFFFFFFFu);                 // global_addr[31:0]
  g0[3] = (unsigned)((ga >> 32) & 0x01FFFFFFu) | (2u << 30);  // addr hi | type=2
  v8i_ g1;
  g1[0] = (int)(1u << 16);                              // data_size=1 (2 bytes)
  g1[1] = (int)((tensor0 & 0xFFFFu) << 16);             // tensor_dim0[15:0]
  g1[2] = (int)(((tensor0 >> 16) & 0xFFFFu) | ((tensor1 & 0xFFFFu) << 16));
  g1[3] = (int)(((tensor1 >> 16) & 0xFFFFu) | ((tile0 & 0xFFFFu) << 16));
  g1[4] = (int)(tile1 & 0xFFFFu);                       // tile_dim1, tile_dim2=0
  g1[5] = (int)(unsigned)(stride0_elems & 0xFFFFFFFFu); // dim0_stride[31:0]
  g1[6] = (int)((stride0_elems >> 32) & 0xFFFFu);       // dim0_stride[47:32]
  g1[7] = 0;
  v4i_ z4 = {0, 0, 0, 0};
  v8i_ z8 = {0, 0, 0, 0, 0, 0, 0, 0};
  __builtin_amdgcn_tensor_load_to_lds(g0, g1, z4, z4, z8, 0);
}
#endif

// ------------- 1) W_embed f32[EMB][F][3] -> bf16 [3][EMB][F] -------------
__global__ __launch_bounds__(256) void conv_w(const float* __restrict__ W,
                                              unsigned short* __restrict__ Wbf) {
  int i = blockIdx.x * 256 + threadIdx.x;          // over EMB*F
  if (i >= EMB_ * F_) return;
  #pragma unroll
  for (int s = 0; s < 3; ++s)
    Wbf[(size_t)s * EMB_ * F_ + i] = f2bf(W[(size_t)i * 3 + s]);
}

// ------------- 2) x f32[B][T][F] -> bf16 xT[B][F][TPAD], zero-padded -------------
__global__ __launch_bounds__(256) void conv_x(const float* __restrict__ x,
                                              unsigned short* __restrict__ xT) {
  size_t i = (size_t)blockIdx.x * 256 + threadIdx.x;   // over B*F*TPAD
  if (i >= (size_t)B_ * F_ * TPAD) return;
  int tp = (int)(i % TPAD);
  size_t r = i / TPAD;
  int f = (int)(r % F_);
  int b = (int)(r / F_);
  int t = tp - 1;                                      // tp=0 is left pad
  float v = (t >= 0 && t < T_) ? x[((size_t)(b * T_ + t)) * F_ + f] : 0.f;
  xT[i] = f2bf(v);
}

// ------------- 3) WMMA GEMM: emb = ReLU(sum_s W_s @ x(shift s) + b) -------------
// Block = 8 waves = 128(M) x 32(N) macro-tile. Double-buffered K-slabs:
// wave 0 waits TENSORcnt for slab i, then issues slab i+1's 6 TDM tiles
// (3 W-tap A slabs + 3 shift-aligned x slabs) into the other buffer while
// all waves run WMMA on slab i from LDS (16-byte ds_load_b128 fragments).
__global__ __launch_bounds__(256) void gemm_emb(
    const unsigned short* __restrict__ Wbf,   // [3][EMB][F] bf16
    const unsigned short* __restrict__ xT,    // [B][F][TPAD] bf16
    const float* __restrict__ b_embed,
    float* __restrict__ emb)                  // [B][T][EMB] f32
{
  extern __shared__ char smem_raw[];

  const int nb   = blockIdx.x;        // N macro tile (32 time cols)
  const int mb   = blockIdx.y;        // M macro tile (128 rows)
  const int b    = blockIdx.z;
  const int wave = threadIdx.x >> 5;
  const int lane = threadIdx.x & 31;
  const int mrow = lane & 15;         // A-matrix row within 16-row tile
  const int half = lane >> 4;         // lane group (ISA 16-bit A layout)

  v8f c0 = {0.f, 0.f, 0.f, 0.f, 0.f, 0.f, 0.f, 0.f};
  v8f c1 = {0.f, 0.f, 0.f, 0.f, 0.f, 0.f, 0.f, 0.f};

#if USE_TDM
  const unsigned ldsBase = (unsigned)(unsigned long long)(void*)smem_raw;
  // prologue: issue slab 0 into buffer 0
  if (wave == 0) {
    #pragma unroll
    for (int s = 0; s < 3; ++s) {
      tdm_load_2d(ldsBase + (unsigned)(s * 128 * KSLAB * 2),
                  Wbf + ((size_t)(s * EMB_ + mb * 128)) * F_,
                  KSLAB, 128, F_, 3 * EMB_, F_);
      tdm_load_2d(ldsBase + (unsigned)(SMEM_A + s * KSLAB * 32 * 2),
                  xT + ((size_t)b * F_) * TPAD + nb * 32 + s,
                  32, KSLAB, TPAD, F_, TPAD);
    }
  }
#endif

  for (int i = 0; i < NSLABS; ++i) {
    const int ks = i * KSLAB;
#if USE_TDM
    if (wave == 0) __builtin_amdgcn_s_wait_tensorcnt(0);  // slab i resident
    __syncthreads();                    // publish slab i; buf i+1 free to fill
    if (wave == 0 && i + 1 < NSLABS) {
      const unsigned nbuf = ldsBase + (unsigned)(((i + 1) & 1) * SMEM_BUF);
      const int ks1 = ks + KSLAB;
      #pragma unroll
      for (int s = 0; s < 3; ++s) {
        tdm_load_2d(nbuf + (unsigned)(s * 128 * KSLAB * 2),
                    Wbf + ((size_t)(s * EMB_ + mb * 128)) * F_ + ks1,
                    KSLAB, 128, F_, 3 * EMB_, F_);
        tdm_load_2d(nbuf + (unsigned)(SMEM_A + s * KSLAB * 32 * 2),
                    xT + ((size_t)(b * F_ + ks1)) * TPAD + nb * 32 + s,
                    32, KSLAB, TPAD, F_, TPAD);
      }
    }
#else
    __syncthreads();                    // previous slab fully consumed
    {
      // cooperative copy fallback (same LDS layout, buffer i&1)
      unsigned short* bufA = (unsigned short*)(smem_raw + (i & 1) * SMEM_BUF);
      unsigned short* bufB = bufA + SMEM_A / 2;
      const unsigned* W32 = (const unsigned*)Wbf;
      unsigned* dA = (unsigned*)bufA;
      for (int idx = threadIdx.x; idx < 3 * 128 * (KSLAB / 2); idx += 256) {
        int s2  = idx / (128 * (KSLAB / 2));
        int rem = idx % (128 * (KSLAB / 2));
        int row = rem / (KSLAB / 2);
        int kd  = rem % (KSLAB / 2);
        dA[idx] = W32[(((size_t)(s2 * EMB_ + mb * 128 + row)) * F_ + ks) / 2 + kd];
      }
      for (int idx = threadIdx.x; idx < 3 * KSLAB * 32; idx += 256) {
        int s2  = idx / (KSLAB * 32);
        int rem = idx % (KSLAB * 32);
        int k   = rem / 32, col = rem % 32;
        bufB[idx] = xT[((size_t)(b * F_ + ks + k)) * TPAD + nb * 32 + s2 + col];
      }
    }
    __syncthreads();
#endif

    const unsigned short* bufA = (const unsigned short*)(smem_raw + (i & 1) * SMEM_BUF);
    const unsigned short* bufB = bufA + SMEM_A / 2;

    for (int s = 0; s < 3; ++s) {
      const unsigned short* aBase = bufA + (size_t)(s * 128 + wave * 16 + mrow) * KSLAB;
      for (int kb = 0; kb < KSLAB; kb += 32) {
        // A fragment: two aligned 16-byte LDS vector loads (ds_load_b128)
        //   dwords 0..3 = K pairs at kb + half*8 + {0,2,4,6}
        //   dwords 4..7 = K pairs at kb + 16 + half*8 + {0,2,4,6}
        v4u a_lo = *(const v4u*)(aBase + kb + half * 8);
        v4u a_hi = *(const v4u*)(aBase + kb + 16 + half * 8);
        v8u a = __builtin_shufflevector(a_lo, a_hi, 0, 1, 2, 3, 4, 5, 6, 7);
        // shift-aligned B slab: column = lane&15, K rows kb + half*8 + {0..7} and kb + 16 + half*8 + {0..7}
        const unsigned short* bRow = bufB + ((size_t)(s * KSLAB) + kb + half * 8) * 32 + (lane & 15);
        #pragma unroll
        for (int n = 0; n < 2; ++n) {
          // B fragment: 16 strided K elements of one column
          unsigned short bt[16];
          #pragma unroll
          for (int j = 0; j < 8; ++j) {
            bt[j]     = bRow[j * 32 + n * 16];
            bt[8 + j] = bRow[(16 + j) * 32 + n * 16];
          }
          v8u bfrag = __builtin_bit_cast(v8u, bt);
          if (n == 0)
            c0 = __builtin_amdgcn_wmma_f32_16x16x32_bf16(
                false, __builtin_bit_cast(v16bf, a),
                false, __builtin_bit_cast(v16bf, bfrag),
                (short)0, c0, false, false);
          else
            c1 = __builtin_amdgcn_wmma_f32_16x16x32_bf16(
                false, __builtin_bit_cast(v16bf, a),
                false, __builtin_bit_cast(v16bf, bfrag),
                (short)0, c1, false, false);
        }
      }
    }
#if USE_TDM
    __syncthreads();                    // all waves done with slab i before reuse
#endif
  }

  // C/D layout: VGPR r -> m = r + 8*half, n = lane&15
  int mbase = mb * 128 + wave * 16 + 8 * half;
  #pragma unroll
  for (int n = 0; n < 2; ++n) {
    int t = nb * 32 + n * 16 + (lane & 15);
    if (t < T_) {
      float* dst = emb + ((size_t)(b * T_ + t)) * EMB_ + mbase;
      #pragma unroll
      for (int r = 0; r < 8; ++r) {
        float cv = (n == 0) ? c0[r] : c1[r];
        dst[r] = fmaxf(cv + b_embed[mbase + r], 0.f);
      }
    }
  }
}

// ------------- 4) cas = emb @ W_cls + b_cls ; act_raw = sum_c cas -------------
__global__ __launch_bounds__(256) void cas_kernel(
    const float* __restrict__ emb, const float* __restrict__ Wcls,
    const float* __restrict__ bcls, float* __restrict__ cas_out,
    float* __restrict__ act_raw)
{
  __shared__ float sred[8];
  __shared__ float scas[C_];
  int bt = blockIdx.x;                       // b*T + t
  const float* e = emb + (size_t)bt * EMB_;
  float er[8];
  #pragma unroll
  for (int j = 0; j < 8; ++j) er[j] = e[threadIdx.x + j * 256];

  for (int cc = 0; cc < C_; ++cc) {
    const float* wr = Wcls + (size_t)cc * EMB_;
    float p = 0.f;
    #pragma unroll
    for (int j = 0; j < 8; ++j) p += er[j] * wr[threadIdx.x + j * 256];
    for (int off = 16; off; off >>= 1) p += __shfl_down(p, off);
    if ((threadIdx.x & 31) == 0) sred[threadIdx.x >> 5] = p;
    __syncthreads();
    if (threadIdx.x == 0) {
      float s2 = 0.f;
      for (int w = 0; w < 8; ++w) s2 += sred[w];
      scas[cc] = s2 + bcls[cc];
    }
    __syncthreads();
  }
  if (threadIdx.x < C_) cas_out[(size_t)bt * C_ + threadIdx.x] = scas[threadIdx.x];
  if (threadIdx.x == 0) {
    float a = 0.f;
    for (int cc = 0; cc < C_; ++cc) a += scas[cc];
    act_raw[bt] = a;
  }
}

// ------------- 5) circular temporal smoothing -------------
__global__ __launch_bounds__(256) void smooth_act(const float* __restrict__ a,
                                                  float* __restrict__ out_act,
                                                  float* __restrict__ act_s) {
  int i = blockIdx.x * 256 + threadIdx.x;
  if (i >= B_ * T_) return;
  int b = i / T_, t = i % T_;
  const float* ab = a + b * T_;
  float v = ab[t] + 0.1f  * (ab[(t + 1) % T_] + ab[(t + T_ - 1) % T_])
                  + 0.02f * (ab[(t + 2) % T_] + ab[(t + T_ - 2) % T_]);
  out_act[i] = v;
  act_s[i]   = v;
}

// ------------- 6) in-LDS bitonic sort (descending, value+index) -------------
__global__ __launch_bounds__(512) void sort_desc_idx(
    const float* __restrict__ scores,   // [B][T]
    int* __restrict__ idx_out,          // [B][1024]
    float* __restrict__ med_out)        // [B] or nullptr
{
  __shared__ float key[1024];
  __shared__ int   sidx[1024];
  int b = blockIdx.x;
  for (int t = threadIdx.x; t < 1024; t += 512) {
    key[t]  = (t < T_) ? scores[b * T_ + t] : -__builtin_inff();
    sidx[t] = t;
  }
  for (int k2 = 2; k2 <= 1024; k2 <<= 1) {
    for (int j = k2 >> 1; j > 0; j >>= 1) {
      __syncthreads();
      for (int t = threadIdx.x; t < 1024; t += 512) {
        int ixj = t ^ j;
        if (ixj > t) {
          bool up = ((t & k2) == 0);
          float a0 = key[t], a1 = key[ixj];
          if (up ? (a0 < a1) : (a0 > a1)) {
            key[t] = a1; key[ixj] = a0;
            int i0 = sidx[t]; sidx[t] = sidx[ixj]; sidx[ixj] = i0;
          }
        }
      }
    }
  }
  __syncthreads();
  for (int t = threadIdx.x; t < 1024; t += 512) idx_out[b * 1024 + t] = sidx[t];
  if (med_out && threadIdx.x == 0)
    med_out[b] = 0.5f * (key[374] + key[375]);   // median of 750 values
}

// ------------- 7) erode/dilate boundary scores -------------
__global__ __launch_bounds__(256) void morph_kernel(
    const float* __restrict__ acts, const float* __restrict__ med,
    float* __restrict__ scB, float* __restrict__ scC) {
  int i = blockIdx.x * 256 + threadIdx.x;
  if (i >= B_ * T_) return;
  int b = i / T_, t = i % T_;
  float m = med[b];
  const float* ab = acts + b * T_;
  float e3 = 1.f, e6 = 1.f, d3 = 0.f, d6 = 0.f;
  #pragma unroll
  for (int dd = -3; dd <= 3; ++dd) {
    int tt = t + dd;
    float v = (tt >= 0 && tt < T_) ? ((ab[tt] > m) ? 1.f : 0.f) : 0.f;
    if (dd >= -1 && dd <= 1) { e3 = fminf(e3, v); d3 = fmaxf(d3, v); }
    if (dd >= -3 && dd <= 2)   e6 = fminf(e6, v);   // erode size 6: [t-3, t+2]
    if (dd >= -2 && dd <= 3)   d6 = fmaxf(d6, v);   // dilate size 6: [t-2, t+3]
  }
  float a = ab[t];
  scB[i] = a * (e3 - e6);   // inner boundary
  scC[i] = a * (d6 - d3);   // outer boundary
}

// ------------- 8) gather embedding rows by sorted indices -------------
__global__ __launch_bounds__(256) void gather_rows(
    const float* __restrict__ emb, const int* __restrict__ idx,
    int reverse, int k, float* __restrict__ out) {
  int j = blockIdx.x, b = blockIdx.y;
  int src = reverse ? idx[b * 1024 + (T_ - 1 - j)] : idx[b * 1024 + j];
  const float* s = emb + ((size_t)(b * T_ + src)) * EMB_;
  float* d = out + ((size_t)(b * k + j)) * EMB_;
  for (int e = threadIdx.x; e < EMB_; e += 256) d[e] = s[e];
}

// ------------- 9) per-class top-150 mean over time (bitonic, keys only) -------------
__global__ __launch_bounds__(512) void topk_mean(const float* __restrict__ cas,
                                                 float* __restrict__ means) {
  __shared__ float key[1024];
  int row = blockIdx.x;                 // b*C + c
  int b = row / C_, c = row % C_;
  for (int t = threadIdx.x; t < 1024; t += 512)
    key[t] = (t < T_) ? cas[((size_t)(b * T_ + t)) * C_ + c] : -__builtin_inff();
  for (int k2 = 2; k2 <= 1024; k2 <<= 1) {
    for (int j = k2 >> 1; j > 0; j >>= 1) {
      __syncthreads();
      for (int t = threadIdx.x; t < 1024; t += 512) {
        int ixj = t ^ j;
        if (ixj > t) {
          bool up = ((t & k2) == 0);
          float a0 = key[t], a1 = key[ixj];
          if (up ? (a0 < a1) : (a0 > a1)) { key[t] = a1; key[ixj] = a0; }
        }
      }
    }
  }
  __syncthreads();
  if (threadIdx.x == 0) {
    float s = 0.f;
    for (int i = 0; i < KEASY; ++i) s += key[i];
    means[row] = s * (1.f / KEASY);
  }
}

// ------------- 10) softmax over classes -------------
__global__ void softmax20(const float* __restrict__ means, float* __restrict__ out) {
  int b = threadIdx.x;
  if (b >= B_) return;
  const float* m = means + b * C_;
  float mx = m[0];
  for (int c = 1; c < C_; ++c) mx = fmaxf(mx, m[c]);
  float e[C_], s = 0.f;
  for (int c = 0; c < C_; ++c) { e[c] = __expf(m[c] - mx); s += e[c]; }
  float inv = 1.f / s;
  for (int c = 0; c < C_; ++c) out[b * C_ + c] = e[c] * inv;
}

// ---------------- launcher ----------------
extern "C" void kernel_launch(void* const* d_in, const int* in_sizes, int n_in,
                              void* d_out, int out_size, void* d_ws, size_t ws_size,
                              hipStream_t stream) {
  (void)in_sizes; (void)n_in; (void)out_size; (void)ws_size;
  const float* x  = (const float*)d_in[0];   // [B][T][F]
  const float* We = (const float*)d_in[1];   // [EMB][F][3]
  const float* be = (const float*)d_in[2];   // [EMB]
  const float* Wc = (const float*)d_in[3];   // [C][EMB][1]
  const float* bc = (const float*)d_in[4];   // [C]
  float* out = (float*)d_out;

  // output layout: concat in reference return order, all f32
  const size_t o_vs  = 0;
  const size_t o_ea  = o_vs + (size_t)B_ * C_;
  const size_t o_eb  = o_ea + (size_t)B_ * KEASY * EMB_;
  const size_t o_ha  = o_eb + (size_t)B_ * KEASY * EMB_;
  const size_t o_hb  = o_ha + (size_t)B_ * KHARD * EMB_;
  const size_t o_act = o_hb + (size_t)B_ * KHARD * EMB_;
  const size_t o_cas = o_act + (size_t)B_ * T_;

  // workspace carve-out
  char* ws = (char*)d_ws;
  size_t off = 0;
  auto take = [&](size_t bytes) -> char* {
    char* p = ws + off;
    off = (off + bytes + 255) & ~(size_t)255;
    return p;
  };
  unsigned short* Wbf = (unsigned short*)take((size_t)3 * EMB_ * F_ * 2);
  unsigned short* xT  = (unsigned short*)take((size_t)B_ * F_ * TPAD * 2);
  float* emb   = (float*)take((size_t)B_ * T_ * EMB_ * 4);
  float* actr  = (float*)take((size_t)B_ * T_ * 4);
  float* acts  = (float*)take((size_t)B_ * T_ * 4);
  int*   idxA  = (int*)take((size_t)B_ * 1024 * 4);
  int*   idxB  = (int*)take((size_t)B_ * 1024 * 4);
  int*   idxC  = (int*)take((size_t)B_ * 1024 * 4);
  float* med   = (float*)take((size_t)B_ * 4);
  float* scB   = (float*)take((size_t)B_ * T_ * 4);
  float* scC   = (float*)take((size_t)B_ * T_ * 4);
  float* means = (float*)take((size_t)B_ * C_ * 4);

  conv_w<<<(EMB_ * F_ + 255) / 256, 256, 0, stream>>>(We, Wbf);
  conv_x<<<(int)(((size_t)B_ * F_ * TPAD + 255) / 256), 256, 0, stream>>>(x, xT);
  // 24 N-macros (32 cols) x 16 M-macros (128 rows) x 32 batches
  gemm_emb<<<dim3(24, 16, 32), 256, SMEM_TOTAL, stream>>>(Wbf, xT, be, emb);
  cas_kernel<<<B_ * T_, 256, 0, stream>>>(emb, Wc, bc, out + o_cas, actr);
  smooth_act<<<(B_ * T_ + 255) / 256, 256, 0, stream>>>(actr, out + o_act, acts);
  sort_desc_idx<<<B_, 512, 0, stream>>>(acts, idxA, med);
  morph_kernel<<<(B_ * T_ + 255) / 256, 256, 0, stream>>>(acts, med, scB, scC);
  sort_desc_idx<<<B_, 512, 0, stream>>>(scB, idxB, nullptr);
  sort_desc_idx<<<B_, 512, 0, stream>>>(scC, idxC, nullptr);
  gather_rows<<<dim3(KEASY, B_), 256, 0, stream>>>(emb, idxA, 0, KEASY, out + o_ea);
  gather_rows<<<dim3(KEASY, B_), 256, 0, stream>>>(emb, idxA, 1, KEASY, out + o_eb);
  gather_rows<<<dim3(KHARD, B_), 256, 0, stream>>>(emb, idxB, 0, KHARD, out + o_ha);
  gather_rows<<<dim3(KHARD, B_), 256, 0, stream>>>(emb, idxC, 0, KHARD, out + o_hb);
  topk_mean<<<B_ * C_, 512, 0, stream>>>(out + o_cas, means);
  softmax20<<<1, 32, 0, stream>>>(means, out + o_vs);
}
